// MaskedHead_35081292873920
// MI455X (gfx1250) — matrix operand order repaired and checked by reading, not compiled, not run on hardware
//
#include <hip/hip_runtime.h>

typedef __attribute__((ext_vector_type(16))) _Float16 v16h;
typedef __attribute__((ext_vector_type(2)))  _Float16 h2;
typedef __attribute__((ext_vector_type(8)))  float    v8f;
typedef __attribute__((ext_vector_type(2)))  float    f2;
typedef __attribute__((ext_vector_type(4)))  unsigned u32x4;
typedef __attribute__((ext_vector_type(8)))  int      i32x8;
typedef __attribute__((ext_vector_type(4)))  int      i32x4;

#define B_  8
#define T_  2048
#define C_  1024
#define HS_ 64
#define BT_ (B_ * T_)
#define CHUNK_ 256              // x K-chunk staged in LDS by the TDM

// ---------------------------------------------------------------------------
// Tensor Data Mover: 2D tile load (global -> LDS), D# built per ISA §8.3/8.4.
// data_size = 4 bytes. Dimensions/strides in elements.
// ---------------------------------------------------------------------------
__device__ __forceinline__ void tdm_load_2d_f32(unsigned lds_addr, const void* gaddr,
                                                unsigned tile_w, unsigned tile_h,
                                                unsigned tensor_w, unsigned tensor_h,
                                                unsigned row_stride) {
  unsigned long long ga = (unsigned long long)gaddr;
  u32x4 g0;
  g0[0] = 1u;                                            // count=1, user D#
  g0[1] = lds_addr;                                      // LDS byte address
  g0[2] = (unsigned)(ga & 0xFFFFFFFFu);                  // global_addr[31:0]
  g0[3] = (unsigned)((ga >> 32) & 0x1FFFFFFu) | (2u << 30); // addr[56:32] | type=2
  i32x8 g1;
  g1[0] = (int)(2u << 16);                               // data_size=4B, no mask/pad
  g1[1] = (int)((tensor_w & 0xFFFFu) << 16);             // tensor_dim0[15:0]
  g1[2] = (int)(((tensor_w >> 16) & 0xFFFFu) | ((tensor_h & 0xFFFFu) << 16));
  g1[3] = (int)(((tensor_h >> 16) & 0xFFFFu) | ((tile_w & 0xFFFFu) << 16));
  g1[4] = (int)(tile_h & 0xFFFFu);                       // tile_dim1 (tile_dim2=0)
  g1[5] = (int)row_stride;                               // tensor_dim0_stride[31:0]
  g1[6] = 0;
  g1[7] = 0;
  i32x4 z4 = {0, 0, 0, 0};
#if __clang_major__ >= 23
  i32x8 z8 = {0, 0, 0, 0, 0, 0, 0, 0};
  __builtin_amdgcn_tensor_load_to_lds(g0, g1, z4, z4, z8, 0);
#else
  __builtin_amdgcn_tensor_load_to_lds(g0, g1, z4, z4, 0);
#endif
}

// 16-lane (half-wave) reductions: xor offsets < 16 stay within each wave32 half,
// matching the C/D layout (lanes 0-15 / 16-31 hold N=0..15 of rows m / m+8).
__device__ __forceinline__ float half_max(float v) {
#pragma unroll
  for (int off = 1; off < 16; off <<= 1) v = fmaxf(v, __shfl_xor(v, off, 32));
  return v;
}
__device__ __forceinline__ float half_sum(float v) {
#pragma unroll
  for (int off = 1; off < 16; off <<= 1) v += __shfl_xor(v, off, 32);
  return v;
}

// ---------------------------------------------------------------------------
// Kernel 0: pre-swizzle Wq/Wk/Wv (fp32 [C,HS]) into f16 B-fragment order.
// Fragment f = (k0/32)*4 + ntile; slot layout: [f][lane][e], e contiguous, so
// kernel 1 fetches each B operand as ONE 32-byte v16h load.
// ---------------------------------------------------------------------------
__global__ __launch_bounds__(256) void wswizzle_kernel(
    const float* __restrict__ Wq, const float* __restrict__ Wk,
    const float* __restrict__ Wv, _Float16* __restrict__ wqf,
    _Float16* __restrict__ wkf, _Float16* __restrict__ wvf) {
  const int idx = blockIdx.x * 256 + threadIdx.x;   // 0 .. C*HS-1
  if (idx >= C_ * HS_) return;
  const int e  = idx & 15;
  const int l  = (idx >> 4) & 31;
  const int f  = idx >> 9;
  const int k  = (f >> 2) * 32 + (l >> 4) * 16 + e; // B-frag: K = h*16 + e
  const int n  = (f & 3) * 16 + (l & 15);
  const size_t src = (size_t)k * HS_ + n;
  wqf[idx] = (_Float16)Wq[src];
  wkf[idx] = (_Float16)Wk[src];
  wvf[idx] = (_Float16)Wv[src];
}

// ---------------------------------------------------------------------------
// Kernel 1: q/k/v projections. x tile (16 x 1024 fp32) is streamed into LDS by
// the TDM in double-buffered 16x256 chunks; A fragments read from LDS (b64),
// B fragments are single v16h loads of the pre-swizzled weights. V is stored
// transposed [B,HS,T] so kernel 2's V fragments are contiguous.
// ---------------------------------------------------------------------------
__global__ __launch_bounds__(128) void qkv_project_kernel(
    const float* __restrict__ x,
    const _Float16* __restrict__ wqf, const _Float16* __restrict__ wkf,
    const _Float16* __restrict__ wvf,
    _Float16* __restrict__ qb, _Float16* __restrict__ kb,
    _Float16* __restrict__ vt) {
  __shared__ float xb[2][16 * CHUNK_];               // 2 x 16 KB
  const int lane = threadIdx.x & 31;
  const int wid  = threadIdx.x >> 5;
  const int m0   = blockIdx.x * 16;                  // row tile in [0, BT)
  const int ml   = lane & 15;
  const int h    = lane >> 4;
  const int n    = wid * 16 + ml;                    // output column (0..63)

  v8f accq = {}, acck = {}, accv = {};

  if (wid == 0) {
    tdm_load_2d_f32((unsigned)(size_t)&xb[0][0], x + (size_t)m0 * C_,
                    CHUNK_, 16, C_, 16, C_);
  }
#pragma unroll
  for (int c = 0; c < C_ / CHUNK_; ++c) {
    if (wid == 0) {
      if (c < C_ / CHUNK_ - 1) {
        tdm_load_2d_f32((unsigned)(size_t)&xb[(c + 1) & 1][0],
                        x + (size_t)m0 * C_ + (c + 1) * CHUNK_,
                        CHUNK_, 16, C_, 16, C_);
        __builtin_amdgcn_s_wait_tensorcnt(1);        // current chunk landed
      } else {
        __builtin_amdgcn_s_wait_tensorcnt(0);        // last chunk landed
      }
    }
    __syncthreads();                                 // LDS chunk visible to all waves
    const float* xc = &xb[c & 1][0];
#pragma unroll
    for (int kk = 0; kk < CHUNK_; kk += 32) {
      // A fragment from LDS: K = h*16 + i*2 (+pair)
      v16h a;
#pragma unroll
      for (int i = 0; i < 8; ++i) {
        const int K = h * 16 + i * 2;
        f2 xv = *(const f2*)(xc + ml * CHUNK_ + kk + K);
        a[2 * i]     = (_Float16)xv.x;
        a[2 * i + 1] = (_Float16)xv.y;
      }
      const int f = (((c * CHUNK_ + kk) >> 5) << 2) + wid;
      const size_t fo = ((size_t)f * 32 + lane) * 16;
      v16h bq = *(const v16h*)(wqf + fo);
      v16h bk = *(const v16h*)(wkf + fo);
      v16h bv = *(const v16h*)(wvf + fo);
      accq = __builtin_amdgcn_wmma_f32_16x16x32_f16(false, a, false, bq, (short)0, accq, false, false);
      acck = __builtin_amdgcn_wmma_f32_16x16x32_f16(false, a, false, bk, (short)0, acck, false, false);
      accv = __builtin_amdgcn_wmma_f32_16x16x32_f16(false, a, false, bv, (short)0, accv, false, false);
    }
    __syncthreads();                                 // done reading before reuse
  }
  // C/D layout: VGPR r holds (row = r + 8h, col = lane&15)
#pragma unroll
  for (int r = 0; r < 8; ++r) {
    const size_t row = (size_t)m0 + r + 8 * h;       // global token row
    qb[row * HS_ + n] = (_Float16)accq[r];
    kb[row * HS_ + n] = (_Float16)acck[r];
    const size_t bb = row >> 11;                     // T = 2048
    const size_t tt = row & (T_ - 1);
    vt[(bb * HS_ + n) * T_ + tt] = (_Float16)accv[r];  // V transposed [B,HS,T]
  }
}

// ---------------------------------------------------------------------------
// Kernel 2: causal flash attention. One 16-query tile per wave, 4 waves/block.
// K^T and V^T B-fragments are single contiguous v16h loads (L2-resident);
// P transposes C-layout -> A-layout through a per-wave LDS slice.
// ---------------------------------------------------------------------------
__global__ __launch_bounds__(128) void attn_kernel(
    const _Float16* __restrict__ qb, const _Float16* __restrict__ kb,
    const _Float16* __restrict__ vt, float* __restrict__ out) {
  __shared__ _Float16 plds[4][16 * 32];              // per-wave 16x32 P tile

  const int lane = threadIdx.x & 31;
  const int wid  = threadIdx.x >> 5;
  const int tile = blockIdx.x * 4 + wid;             // global 16-query tile id
  const int b    = tile >> 7;                        // T/16 = 128 tiles/batch
  const int q0   = (tile & 127) * 16;
  const int ml   = lane & 15;
  const int h    = lane >> 4;
  _Float16* pl   = plds[wid];

  // Q A-fragments for the two HS chunks (K = 0..31 and 32..63)
  v16h a0, a1;
  const _Float16* qrow = qb + (size_t)(b * T_ + q0 + ml) * HS_;
#pragma unroll
  for (int i = 0; i < 8; ++i) {
    const int K = h * 16 + i * 2;
    h2 p0 = *(const h2*)(qrow + K);
    h2 p1 = *(const h2*)(qrow + 32 + K);
    a0[2 * i] = p0.x; a0[2 * i + 1] = p0.y;
    a1[2 * i] = p1.x; a1[2 * i + 1] = p1.y;
  }

  float mrow[8], lrow[8];
#pragma unroll
  for (int r = 0; r < 8; ++r) { mrow[r] = -1e30f; lrow[r] = 0.0f; }
  v8f acc[4] = {};                                   // 16x64 output accumulator

  const float scale = 0.03125f;                      // C^-0.5 = 1/32

  for (int s0 = 0; s0 < q0 + 16; s0 += 32) {
    if (s0 + 32 < q0 + 16) {                         // global_prefetch next K tile
      __builtin_prefetch(kb + (size_t)(b * T_ + s0 + 32 + ml) * HS_, 0, 1);
    }
    // --- S = (Q @ K^T) * scale : two 16x16 j-subtiles ---
    v8f st[2];
#pragma unroll
    for (int jt = 0; jt < 2; ++jt) {
      const int key = s0 + jt * 16 + ml;
      const _Float16* krow = kb + (size_t)(b * T_ + key) * HS_ + h * 16;
      v16h kf0 = *(const v16h*)(krow);
      v16h kf1 = *(const v16h*)(krow + 32);
      v8f s = {};
      s = __builtin_amdgcn_wmma_f32_16x16x32_f16(false, a0, false, kf0, (short)0, s, false, false);
      s = __builtin_amdgcn_wmma_f32_16x16x32_f16(false, a1, false, kf1, (short)0, s, false, false);
      st[jt] = s;
    }
    // scale + causal mask (elementwise, no divergence)
#pragma unroll
    for (int jt = 0; jt < 2; ++jt) {
      const int kj = s0 + jt * 16 + ml;
#pragma unroll
      for (int r = 0; r < 8; ++r) {
        const int qi = q0 + r + 8 * h;
        st[jt][r] = (kj > qi) ? -1e30f : st[jt][r] * scale;
      }
    }
    // --- online softmax update (per row r+8h) ---
    float alpha[8];
#pragma unroll
    for (int r = 0; r < 8; ++r) {
      float tm   = half_max(fmaxf(st[0][r], st[1][r]));
      float mnew = fmaxf(mrow[r], tm);
      alpha[r]   = __expf(mrow[r] - mnew);
      mrow[r]    = mnew;
      st[0][r]   = __expf(st[0][r] - mnew);
      st[1][r]   = __expf(st[1][r] - mnew);
      lrow[r]    = lrow[r] * alpha[r] + half_sum(st[0][r] + st[1][r]);
    }
#pragma unroll
    for (int t = 0; t < 4; ++t)
#pragma unroll
      for (int r = 0; r < 8; ++r) acc[t][r] *= alpha[r];

    // --- P: C layout -> per-wave LDS -> A-fragment layout (in-order DS) ---
#pragma unroll
    for (int jt = 0; jt < 2; ++jt)
#pragma unroll
      for (int r = 0; r < 8; ++r)
        pl[(r + 8 * h) * 32 + jt * 16 + ml] = (_Float16)st[jt][r];

    v16h pa;
#pragma unroll
    for (int i = 0; i < 8; ++i) {
      const int K = h * 16 + i * 2;
      h2 pp = *(const h2*)(pl + ml * 32 + K);
      pa[2 * i] = pp.x; pa[2 * i + 1] = pp.y;
    }
    // --- O += P @ V : V^T fragment = contiguous v16h (K runs along keys) ---
#pragma unroll
    for (int t = 0; t < 4; ++t) {
      const _Float16* vrow =
          vt + ((size_t)b * HS_ + t * 16 + ml) * T_ + s0 + h * 16;
      v16h vf = *(const v16h*)vrow;
      acc[t] = __builtin_amdgcn_wmma_f32_16x16x32_f16(false, pa, false, vf, (short)0, acc[t], false, false);
    }
  }
  // epilogue: normalize by row sum, write fp32
#pragma unroll
  for (int t = 0; t < 4; ++t)
#pragma unroll
    for (int r = 0; r < 8; ++r) {
      const size_t row = (size_t)b * T_ + q0 + r + 8 * h;
      out[row * HS_ + t * 16 + ml] = acc[t][r] / lrow[r];
    }
}

extern "C" void kernel_launch(void* const* d_in, const int* in_sizes, int n_in,
                              void* d_out, int out_size, void* d_ws, size_t ws_size,
                              hipStream_t stream) {
  const float* x  = (const float*)d_in[0];
  const float* Wq = (const float*)d_in[1];
  const float* Wk = (const float*)d_in[2];
  const float* Wv = (const float*)d_in[3];
  float* out = (float*)d_out;

  // workspace layout (f16): swizzled weights (3 x 128KB), q, k, v^T (3 x 2MB)
  _Float16* wqf = (_Float16*)d_ws;
  _Float16* wkf = wqf + (size_t)C_ * HS_;
  _Float16* wvf = wkf + (size_t)C_ * HS_;
  _Float16* qbb = wvf + (size_t)C_ * HS_;
  _Float16* kbb = qbb + (size_t)BT_ * HS_;
  _Float16* vtt = kbb + (size_t)BT_ * HS_;

  wswizzle_kernel<<<(C_ * HS_ + 255) / 256, 256, 0, stream>>>(Wq, Wk, Wv, wqf, wkf, wvf);
  qkv_project_kernel<<<BT_ / 16, 128, 0, stream>>>(x, wqf, wkf, wvf, qbb, kbb, vtt);
  attn_kernel<<<BT_ / 16 / 4, 128, 0, stream>>>(qbb, kbb, vtt, out);
}
